// LocalAttention_27668179321047
// MI455X (gfx1250) — matrix operand order repaired and checked by reading, not compiled, not run on hardware
//
#include <hip/hip_runtime.h>

// ---------------- shapes ----------------
#define Bb 16
#define Cc 1024
#define Nn 32
#define Hh 8
#define Dh 16
#define Dd 128           // H*Dh
#define CB 2             // c-values per workgroup
#define MROWS (CB*Nn)    // 64 gated-neighbor rows per workgroup

typedef __attribute__((ext_vector_type(16))) __bf16 v16bf;
typedef __attribute__((ext_vector_type(8)))  __bf16 v8bf;
typedef __attribute__((ext_vector_type(8)))  float  v8f;

__device__ __forceinline__ float swishf(float x) {
    return x / (1.0f + __expf(-x));   // x * sigmoid(x)
}

// ---------------------------------------------------------------------------
// Prep: re-layout Wk/Wv (fp32 [128,128], row-major [K][N]) into bf16 WMMA
// B-operand fragments: frag[mat][ks][nt][lane][i], K = ks*32 + 16*(lane>>4) + i,
// N = nt*16 + (lane&15).  32768 bf16 total (64 KB in d_ws).
// ---------------------------------------------------------------------------
__global__ __launch_bounds__(256) void la_prep(const float* __restrict__ Wk,
                                               const float* __restrict__ Wv,
                                               __bf16* __restrict__ frag) {
    int idx  = blockIdx.x * 256 + threadIdx.x;     // 0 .. 32767
    int mat  = idx >> 14;
    int r    = idx & 16383;
    int ks   = r >> 12;
    int nt   = (r >> 9) & 7;
    int lane = (r >> 4) & 31;
    int i    = r & 15;
    int K    = ks * 32 + ((lane >> 4) << 4) + i;
    int Nc   = nt * 16 + (lane & 15);
    const float* W = mat ? Wv : Wk;
    frag[idx] = (__bf16)W[K * Dd + Nc];
}

// ---------------------------------------------------------------------------
// Main fused kernel: one workgroup per (b, pair-of-c).
// ---------------------------------------------------------------------------
__global__ __launch_bounds__(256) void la_main(
    const float* __restrict__ aq,    // [B,C,D]
    const float* __restrict__ an,    // [B,C,N,D]
    const float* __restrict__ ldist, // [B,C,N,1]
    const float* __restrict__ msk,   // [B,C,N]
    const float* __restrict__ Wq, const float* __restrict__ bq,
    const float* __restrict__ bk, const float* __restrict__ bv,
    const float* __restrict__ Wf, const float* __restrict__ bfv,
    const __bf16* __restrict__ fragK, const __bf16* __restrict__ fragV,
    float* __restrict__ attn_out,    // [B,H,C,N]
    float* __restrict__ ctx_out)     // [B,C,D]
{
    __shared__ __align__(16) __bf16 sAK[MROWS][136]; // gated neighbors, then K (bf16)
    __shared__ float sV[MROWS][132];                 // V projection (fp32)
    __shared__ float sQ[CB][Dd];                     // query projection (fp32)
    __shared__ float sW[CB][Hh][Nn];                 // attn * mask weights

    const int tid  = threadIdx.x;
    const int lane = tid & 31;
    const int wave = tid >> 5;
    const int bid  = blockIdx.x;
    const int b    = bid >> 9;            // / (C/CB = 512)
    const int c0   = (bid & 511) * CB;

    // ---------- Phase 1a: swish-gated neighbors -> sAK (bf16), coalesced ----
    #pragma unroll
    for (int it = 0; it < 8; ++it) {
        int idx = it * 256 + tid;         // 2048 float4 = 64 rows * 32 float4
        int row = idx >> 5;               // 0..63 (one wave covers one row)
        int c4  = (idx & 31) << 2;        // column (float4 granularity)
        int cl  = row >> 5;
        int nb  = row & 31;
        long g  = (long)(b * Cc + c0 + cl) * Nn + nb;
        float dist = ldist[g];
        float4 x  = *(const float4*)(an + g * (long)Dd + c4);
        float4 wf = *(const float4*)(Wf + c4);
        float4 bb = *(const float4*)(bfv + c4);
        sAK[row][c4 + 0] = (__bf16)(x.x * swishf(dist * wf.x + bb.x));
        sAK[row][c4 + 1] = (__bf16)(x.y * swishf(dist * wf.y + bb.y));
        sAK[row][c4 + 2] = (__bf16)(x.z * swishf(dist * wf.z + bb.z));
        sAK[row][c4 + 3] = (__bf16)(x.w * swishf(dist * wf.w + bb.w));
    }

    // ---------- Phase 1b: query = aq @ Wq + bq (fp32 VALU, 2x128 outputs) ---
    {
        int cl = tid >> 7;
        int dq = tid & 127;
        const float* q = aq + (long)(b * Cc + c0 + cl) * Dd;
        float acc = bq[dq];
        #pragma unroll 4
        for (int k = 0; k < Dd; ++k) acc += q[k] * Wq[k * Dd + dq];
        sQ[cl][dq] = acc;
    }
    __syncthreads();

    // ---------- Phase 2: K = an@Wk+bk, V = an@Wv+bv via WMMA bf16 -----------
    // wave -> (mt = wave&3 : 16-row strip, nh = wave>>2 : 4 of 8 N-tiles)
    const int mt = wave & 3;
    const int nh = wave >> 2;
    v8f accK[4], accV[4];
    #pragma unroll
    for (int j = 0; j < 4; ++j) { accK[j] = (v8f)0.0f; accV[j] = (v8f)0.0f; }

    const int arow  = mt * 16 + (lane & 15);
    const int khalf = (lane >> 4) << 4;     // 0 or 16

    #pragma unroll
    for (int ks = 0; ks < 4; ++ks) {
        // A fragment, contiguous K order: elem i -> K=kb+i (kb includes 16*(lane>>4))
        int kb = ks * 32 + khalf;
        v8bf lo = *(const v8bf*)&sAK[arow][kb];
        v8bf hi = *(const v8bf*)&sAK[arow][kb + 8];
        v16bf a;
        #pragma unroll
        for (int i = 0; i < 8; ++i) { a[i] = lo[i]; a[i + 8] = hi[i]; }
        #pragma unroll
        for (int j = 0; j < 4; ++j) {
            int nt = nh * 4 + j;
            long fo = ((long)(ks * 8 + nt) * 32 + lane) * 16;   // 32B/lane, contiguous
            v16bf bK = *(const v16bf*)(fragK + fo);
            v16bf bV = *(const v16bf*)(fragV + fo);
            accK[j] = __builtin_amdgcn_wmma_f32_16x16x32_bf16(
                          false, a, false, bK, (short)0, accK[j], false, false);
            accV[j] = __builtin_amdgcn_wmma_f32_16x16x32_bf16(
                          false, a, false, bV, (short)0, accV[j], false, false);
        }
    }
    __syncthreads();   // all A-fragment reads done before sAK is overwritten with K

    // C/D layout: elem e -> row = mt*16 + e + 8*(lane>>4), col = nt*16 + (lane&15)
    #pragma unroll
    for (int j = 0; j < 4; ++j) {
        int nt  = nh * 4 + j;
        int col = nt * 16 + (lane & 15);
        float bkc = bk[col];
        float bvc = bv[col];
        int rbase = mt * 16 + ((lane >> 4) << 3);
        #pragma unroll
        for (int e = 0; e < 8; ++e) {
            sAK[rbase + e][col] = (__bf16)(accK[j][e] + bkc);
            sV [rbase + e][col] = accV[j][e] + bvc;
        }
    }
    __syncthreads();

    // ---------- Phase 3a: energy + masked softmax (wave32, lane = neighbor) -
    #pragma unroll
    for (int pp = 0; pp < 2; ++pp) {
        int p  = wave * 2 + pp;          // 16 (c,h) pairs over 8 waves
        int cl = p >> 3;
        int h  = p & 7;
        int n  = lane;
        int row = cl * 32 + n;
        float e = 0.0f;
        #pragma unroll
        for (int d = 0; d < Dh; ++d)
            e += sQ[cl][h * Dh + d] * (float)sAK[row][h * Dh + d];
        e *= 0.25f;                      // Dh^-0.5
        long g  = (long)(b * Cc + c0 + cl) * Nn + n;
        float m = msk[g];
        e += (1.0f - m) * (-1e9f);
        float mx = e;
        #pragma unroll
        for (int off = 16; off; off >>= 1) mx = fmaxf(mx, __shfl_xor(mx, off, 32));
        float ex = __expf(e - mx);
        float sm = ex;
        #pragma unroll
        for (int off = 16; off; off >>= 1) sm += __shfl_xor(sm, off, 32);
        float at = ex / sm;
        attn_out[((long)(b * Hh + h) * Cc + (c0 + cl)) * Nn + n] = at;
        sW[cl][h][n] = at * m;
    }
    __syncthreads();

    // ---------- Phase 3b: context = sum_n w * V + query ---------------------
    {
        int cl = tid >> 7;
        int dq = tid & 127;
        int h  = dq >> 4;
        float acc = sQ[cl][dq];
        #pragma unroll
        for (int n = 0; n < Nn; ++n)
            acc += sW[cl][h][n] * sV[cl * 32 + n][dq];
        ctx_out[(long)(b * Cc + c0 + cl) * Dd + dq] = acc;
    }
}

// ---------------------------------------------------------------------------
extern "C" void kernel_launch(void* const* d_in, const int* in_sizes, int n_in,
                              void* d_out, int out_size, void* d_ws, size_t ws_size,
                              hipStream_t stream) {
    const float* aq = (const float*)d_in[0];   // atom_query
    const float* an = (const float*)d_in[1];   // atom_neighbor
    const float* ld = (const float*)d_in[2];   // local_distance
    const float* mk = (const float*)d_in[3];   // mask
    const float* Wq = (const float*)d_in[4];
    const float* bq = (const float*)d_in[5];
    const float* Wk = (const float*)d_in[6];
    const float* bk = (const float*)d_in[7];
    const float* Wv = (const float*)d_in[8];
    const float* bv = (const float*)d_in[9];
    const float* Wf = (const float*)d_in[10];
    const float* bf = (const float*)d_in[11];

    __bf16* frag = (__bf16*)d_ws;                      // 64 KB: Wk frags then Wv frags
    float* attn = (float*)d_out;                       // [B,H,C,N]
    float* ctx  = attn + (long)Bb * Hh * Cc * Nn;      // [B,C,D]

    la_prep<<<128, 256, 0, stream>>>(Wk, Wv, frag);
    la_main<<<Bb * (Cc / CB), 256, 0, stream>>>(
        aq, an, ld, mk, Wq, bq, bk, bv, Wf, bf,
        frag, frag + 16384, attn, ctx);
}
